// SetAbstraction_12601434047023
// MI455X (gfx1250) — matrix operand order repaired and checked by reading, not compiled, not run on hardware
//
#include <hip/hip_runtime.h>
#include <hip/hip_bf16.h>

// ---------------------------------------------------------------------------
// PointNet++ SetAbstraction for MI455X (gfx1250, wave32, WMMA).
// GEMMs use v_wmma_f32_16x16x32_f16 (fp16 in, f32 accumulate), register-
// blocked over N so the activation buffer is streamed exactly once per layer.
// ---------------------------------------------------------------------------

typedef __attribute__((ext_vector_type(16))) _Float16 v16h;
typedef __attribute__((ext_vector_type(8)))  _Float16 v8h;
typedef __attribute__((ext_vector_type(8)))  float    v8f;

#define NB      16
#define NPTS    4096
#define NPOINT  1024
#define NSAMPLE 32
#define NGROUP  (NB * NPOINT)            // 16384
#define MROWS   (NGROUP * NSAMPLE)       // 524288
#define RAD2    0.04f
#define EPSBN   1e-5f

// ---- workspace layout (bytes, all 256-aligned) ----------------------------
#define OFF_NEWXYZ 0u           // [B,S,3] f32                 196608
#define OFF_BALL   196608u      // [B,S,32] i32               2097152
#define OFF_STATS  2293760u     // 3 layers x (sum256,sq256)     6144
#define OFF_AB     2299904u     // 3 layers x (a256,b256)        6144
#define OFF_W1F    2306048u     // L1 B-frags fp16              24576
#define OFF_W2F    2330624u     // L2 B-frags fp16              32768
#define OFF_W3F    2363392u     // L3 B-frags fp16              65536
#define OFF_POOL   2428928u     // [G,256] f32               16777216
#define OFF_BUFA   19206144u    // fp16 act buffer          134217728
#define OFF_BUFB   153423872u   // fp16 act buffer          134217728

// ---------------------------------------------------------------------------
// A-fragment loader: 16x32 fp16 A tile from row-major [M x ldk] fp16 buffer.
// ISA 7.12.2 16-bit A layout: lane row = lane&15; halves 0..7 -> k = base +
// (lane<16?0:8) + i ; halves 8..15 -> +16.  Two 16B loads per lane.
// ---------------------------------------------------------------------------
union AFragU { v16h v; v8h h[2]; };

__device__ __forceinline__ v16h load_a_frag(const _Float16* __restrict__ act,
                                            int ldk, int mbase, int kbase, int lane) {
  int row = mbase + (lane & 15);
  int ko  = kbase + ((lane < 16) ? 0 : 8);
  const _Float16* p = act + (size_t)row * ldk + ko;
  AFragU u;
  u.h[0] = *(const v8h*)(p);
  u.h[1] = *(const v8h*)(p + 16);
  return u.v;
}

__device__ __forceinline__ v8f wmma_f16(v16h a, v16h b, v8f c) {
  return __builtin_amdgcn_wmma_f32_16x16x32_f16(false, a, false, b, (short)0, c,
                                                false, false);
}

// ---------------------------------------------------------------------------
// 0) zero the BN stats accumulators (harness does not re-poison between runs)
// ---------------------------------------------------------------------------
__global__ void zero_stats_kernel(float* __restrict__ p) {
  p[blockIdx.x * 256 + threadIdx.x] = 0.0f;   // grid 6 -> 1536 floats
}

// ---------------------------------------------------------------------------
// 1) Farthest point sampling. One block per batch, xyz in LDS, dist in regs.
//    Writes new_xyz scratch [B,S,3] and the transposed new_xyz output [B,3,S].
// ---------------------------------------------------------------------------
__global__ __launch_bounds__(1024)
void fps_kernel(const float* __restrict__ xyz, float* __restrict__ newxyz,
                float* __restrict__ out_newxyz) {
  const int b = blockIdx.x, tid = threadIdx.x;
  const float* bx = xyz + (size_t)b * 3 * NPTS;
  __shared__ float sx[NPTS], sy[NPTS], sz[NPTS];
  __shared__ float rval[32];
  __shared__ int   ridx[32];
  __shared__ int   sfar;
  for (int i = tid; i < NPTS; i += 1024) {
    sx[i] = bx[i]; sy[i] = bx[NPTS + i]; sz[i] = bx[2 * NPTS + i];
  }
  __syncthreads();
  float px[4], py[4], pz[4], dist[4];
  for (int j = 0; j < 4; ++j) {
    int i = tid * 4 + j;
    px[j] = sx[i]; py[j] = sy[i]; pz[j] = sz[i];
    dist[j] = 1e10f;
  }
  int far = 0;
  const int wv = tid >> 5, ln = tid & 31;
  for (int it = 0; it < NPOINT; ++it) {
    float cx = sx[far], cy = sy[far], cz = sz[far];
    if (tid == 0) {
      size_t g = (size_t)b * NPOINT + it;
      newxyz[g * 3 + 0] = cx; newxyz[g * 3 + 1] = cy; newxyz[g * 3 + 2] = cz;
      size_t o = (size_t)b * 3 * NPOINT + it;
      out_newxyz[o] = cx; out_newxyz[o + NPOINT] = cy; out_newxyz[o + 2 * NPOINT] = cz;
    }
    float bestv = -1.0f; int besti = 0;
    for (int j = 0; j < 4; ++j) {
      float dx = px[j] - cx, dy = py[j] - cy, dz = pz[j] - cz;
      float d  = dx * dx + dy * dy + dz * dz;
      float nd = fminf(dist[j], d);
      dist[j] = nd;
      if (nd > bestv) { bestv = nd; besti = tid * 4 + j; }
    }
    // wave32 argmax (ties -> lowest index, matching jnp.argmax)
    for (int off = 16; off > 0; off >>= 1) {
      float ov = __shfl_xor(bestv, off, 32);
      int   oi = __shfl_xor(besti, off, 32);
      if (ov > bestv || (ov == bestv && oi < besti)) { bestv = ov; besti = oi; }
    }
    if (ln == 0) { rval[wv] = bestv; ridx[wv] = besti; }
    __syncthreads();
    if (wv == 0) {
      float v = rval[ln]; int ii = ridx[ln];
      for (int off = 16; off > 0; off >>= 1) {
        float ov = __shfl_xor(v, off, 32);
        int   oi = __shfl_xor(ii, off, 32);
        if (ov > v || (ov == v && oi < ii)) { v = ov; ii = oi; }
      }
      if (ln == 0) sfar = ii;
    }
    __syncthreads();
    far = sfar;
  }
}

// ---------------------------------------------------------------------------
// 2) Ball query: first 32 in-ball indices in ascending order (== jnp.sort
//    of masked arange truncated to nsample). One thread per query; batch xyz
//    staged in LDS. 4 blocks per batch.
// ---------------------------------------------------------------------------
__global__ __launch_bounds__(256)
void ballquery_kernel(const float* __restrict__ xyz,
                      const float* __restrict__ newxyz,
                      int* __restrict__ ballidx) {
  const int blk = blockIdx.x;
  const int b = blk >> 2, sbase = (blk & 3) * 256;
  __shared__ float sx[NPTS], sy[NPTS], sz[NPTS];
  const float* bx = xyz + (size_t)b * 3 * NPTS;
  for (int i = threadIdx.x; i < NPTS; i += 256) {
    sx[i] = bx[i]; sy[i] = bx[NPTS + i]; sz[i] = bx[2 * NPTS + i];
  }
  __syncthreads();
  const int s = sbase + threadIdx.x;
  const float* q = newxyz + ((size_t)b * NPOINT + s) * 3;
  const float qx = q[0], qy = q[1], qz = q[2];
  int* out = ballidx + ((size_t)b * NPOINT + s) * NSAMPLE;
  int cnt = 0, first = 0;
  for (int i = 0; i < NPTS && cnt < NSAMPLE; ++i) {
    float dx = sx[i] - qx, dy = sy[i] - qy, dz = sz[i] - qz;
    if (dx * dx + dy * dy + dz * dz <= RAD2) {
      if (cnt == 0) first = i;
      out[cnt++] = i;
    }
  }
  for (int j = cnt; j < NSAMPLE; ++j) out[j] = first;  // pad with first hit
}

// ---------------------------------------------------------------------------
// 3) Gather grouped (xyz-centered ++ feats) into fp16 X0 [M x 96] (67 real,
//    zero-padded so K is a multiple of 32 for the WMMA reduction).
// ---------------------------------------------------------------------------
__global__ __launch_bounds__(256)
void build_x0_kernel(const float* __restrict__ xyz, const float* __restrict__ feats,
                     const float* __restrict__ newxyz, const int* __restrict__ ballidx,
                     _Float16* __restrict__ x0) {
  const size_t m = (size_t)blockIdx.x * 256 + threadIdx.x;   // < MROWS
  const int g = (int)(m >> 5);
  const int b = g >> 10;
  const int pi = ballidx[m];
  const float* bx = xyz + (size_t)b * 3 * NPTS;
  const float* nz = newxyz + (size_t)g * 3;
  _Float16* row = x0 + m * 96;
  row[0] = (_Float16)(bx[pi] - nz[0]);
  row[1] = (_Float16)(bx[NPTS + pi] - nz[1]);
  row[2] = (_Float16)(bx[2 * NPTS + pi] - nz[2]);
  const float* bf = feats + (size_t)b * 64 * NPTS;
  for (int c = 0; c < 64; ++c) row[3 + c] = (_Float16)bf[(size_t)c * NPTS + pi];
  for (int c = 67; c < 96; ++c) row[c] = (_Float16)0.0f;
}

// ---------------------------------------------------------------------------
// 4) Pack a weight matrix W[nout x kreal] (f32) into WMMA B-fragments (fp16),
//    zero-padding k >= kreal. ISA 7.12.2 16-bit B layout (32x16 tile):
//    lane n = lane&15, halves i<8 -> k = kc*32 + (lane<16 ? 0 : 8) + i; i>=8 -> +8.
//    Storage: frag[(nt*KCH + kc)*32 + lane][16 halves].
// ---------------------------------------------------------------------------
__global__ void prep_wfrag_kernel(const float* __restrict__ W, int kreal, int kch,
                                  _Float16* __restrict__ frag) {
  const int blk = blockIdx.x;
  const int kc = blk % kch, nt = blk / kch;
  const int lane = threadIdx.x;
  const int n = nt * 16 + (lane & 15);
  const int kb = kc * 32 + ((lane < 16) ? 0 : 8);
  _Float16* dst = frag + ((size_t)(nt * kch + kc) * 32 + lane) * 16;
  for (int i = 0; i < 16; ++i) {
    int k = kb + ((i < 8) ? i : i + 8);
    dst[i] = (_Float16)((k < kreal) ? W[(size_t)n * kreal + k] : 0.0f);
  }
}

// ---------------------------------------------------------------------------
// 5) GEMM layers 1/2 (NOUT=128): one wave owns a full 16x128 output strip
//    (8 accumulators). A is loaded ONCE per K-chunk and reused for all 8
//    column tiles -> activation buffer is streamed a single time per layer.
//    Writes pre-BN fp16 [M x 128] + per-channel sum/sumsq (LDS -> 1 global
//    atomic per channel per block).
// ---------------------------------------------------------------------------
__global__ __launch_bounds__(256)
void gemm_bn_kernel(const _Float16* __restrict__ act, int ldk, int kch,
                    const _Float16* __restrict__ wfrag,
                    _Float16* __restrict__ outbuf,
                    float* __restrict__ gsum, float* __restrict__ gsq) {
  __shared__ float lsum[128], lsq[128];
  const int tid = threadIdx.x;
  if (tid < 128) { lsum[tid] = 0.0f; lsq[tid] = 0.0f; }
  __syncthreads();
  const int lane  = tid & 31;
  const int mtile = blockIdx.x * 8 + (tid >> 5);
  const int mbase = mtile * 16;

  v8f acc[8] = {};
  for (int kc = 0; kc < kch; ++kc) {
    v16h a = load_a_frag(act, ldk, mbase, kc * 32, lane);
#pragma unroll
    for (int nt = 0; nt < 8; ++nt) {
      v16h bf = *(const v16h*)(wfrag + ((size_t)(nt * kch + kc) * 32 + lane) * 16);
      acc[nt] = wmma_f16(a, bf, acc[nt]);
    }
  }
  // D layout: vgpr v, lanes 0-15 -> row v, lanes 16-31 -> row v+8; col = lane&15
  const int cbase = lane & 15;
  const int rbase = mbase + ((lane < 16) ? 0 : 8);
#pragma unroll
  for (int nt = 0; nt < 8; ++nt) {
    const int col = nt * 16 + cbase;
    float s = 0.0f, sq = 0.0f;
#pragma unroll
    for (int v = 0; v < 8; ++v) {
      float x = acc[nt][v];
      s += x; sq += x * x;
      outbuf[(size_t)(rbase + v) * 128 + col] = (_Float16)x;
    }
    s  += __shfl_xor(s, 16, 32);
    sq += __shfl_xor(sq, 16, 32);
    if (lane < 16) { atomicAdd(&lsum[col], s); atomicAdd(&lsq[col], sq); }
  }
  __syncthreads();
  if (tid < 128) { atomicAdd(&gsum[tid], lsum[tid]); atomicAdd(&gsq[tid], lsq[tid]); }
}

// ---------------------------------------------------------------------------
// 6) BN finalize: a = gamma*rsqrt(var+eps), b = beta - a*mean.
//    (conv bias cancels exactly under batch-norm -> omitted.)
// ---------------------------------------------------------------------------
__global__ void bn_final_kernel(const float* __restrict__ gsum, const float* __restrict__ gsq,
                                const float* __restrict__ gamma, const float* __restrict__ beta,
                                float* __restrict__ a, float* __restrict__ bb,
                                int nout, float invM) {
  int c = threadIdx.x;
  if (c >= nout) return;
  float mean = gsum[c] * invM;
  float var  = gsq[c] * invM - mean * mean;
  float ac   = gamma[c] * rsqrtf(var + EPSBN);
  a[c] = ac;
  bb[c] = beta[c] - mean * ac;
}

// ---------------------------------------------------------------------------
// 7) Apply BN + ReLU in place on fp16 [M x 128] (8 channels per thread).
// ---------------------------------------------------------------------------
__global__ __launch_bounds__(256)
void bn_apply_kernel(_Float16* __restrict__ buf, const float* __restrict__ a,
                     const float* __restrict__ bb) {
  const size_t idx = ((size_t)blockIdx.x * 256 + threadIdx.x) * 8;
  const int c0 = (int)(idx & 127);
  v8h v = *(v8h*)(buf + idx);
  for (int j = 0; j < 8; ++j) {
    float y = a[c0 + j] * (float)v[j] + bb[c0 + j];
    v[j] = (_Float16)fmaxf(y, 0.0f);
  }
  *(v8h*)(buf + idx) = v;
}

// ---------------------------------------------------------------------------
// 8) Layer 3 GEMM (K=128, NOUT=256) + stats + fused max-pool over the 32
//    rows of each group (BN+ReLU are monotone since gamma>0, so pooling
//    pre-BN values is exact). One wave owns a full group x 64 columns
//    (2 row tiles x 4 column tiles); each A fragment is reused 4x.
// ---------------------------------------------------------------------------
__global__ __launch_bounds__(256)
void gemm3_pool_kernel(const _Float16* __restrict__ act,
                       const _Float16* __restrict__ wfrag,
                       float* __restrict__ pooled,
                       float* __restrict__ gsum, float* __restrict__ gsq) {
  __shared__ float lsum[256], lsq[256];
  const int tid = threadIdx.x;
  lsum[tid] = 0.0f; lsq[tid] = 0.0f;
  __syncthreads();
  const int lane = tid & 31;
  const int w = blockIdx.x * 8 + (tid >> 5);
  const int ntq = w & 3;               // which quarter of the 16 column tiles
  const int g   = w >> 2;              // group id
  const int mbase = g * 32;

  v8f acc0[4] = {};
  v8f acc1[4] = {};
  for (int kc = 0; kc < 4; ++kc) {
    v16h a0 = load_a_frag(act, 128, mbase,      kc * 32, lane);
    v16h a1 = load_a_frag(act, 128, mbase + 16, kc * 32, lane);
#pragma unroll
    for (int j = 0; j < 4; ++j) {
      const int ntile = ntq * 4 + j;
      v16h bf = *(const v16h*)(wfrag + ((size_t)(ntile * 4 + kc) * 32 + lane) * 16);
      acc0[j] = wmma_f16(a0, bf, acc0[j]);
      acc1[j] = wmma_f16(a1, bf, acc1[j]);
    }
  }
  const int cbase = lane & 15;
#pragma unroll
  for (int j = 0; j < 4; ++j) {
    float s = 0.0f, sq = 0.0f, mx = -3.4e38f;
#pragma unroll
    for (int v = 0; v < 8; ++v) {
      float x0 = acc0[j][v], x1 = acc1[j][v];
      s += x0 + x1; sq += x0 * x0 + x1 * x1;
      mx = fmaxf(mx, fmaxf(x0, x1));
    }
    s  += __shfl_xor(s, 16, 32);
    sq += __shfl_xor(sq, 16, 32);
    mx  = fmaxf(mx, __shfl_xor(mx, 16, 32));   // combine row halves -> 32-row max
    const int col = (ntq * 4 + j) * 16 + cbase;
    if (lane < 16) {
      pooled[(size_t)g * 256 + col] = mx;
      atomicAdd(&lsum[col], s); atomicAdd(&lsq[col], sq);
    }
  }
  __syncthreads();
  atomicAdd(&gsum[tid], lsum[tid]); atomicAdd(&gsq[tid], lsq[tid]);
}

// ---------------------------------------------------------------------------
// 9) Final: relu(a*pooled+b) -> new_points output [B,256,S] after new_xyz.
// ---------------------------------------------------------------------------
__global__ __launch_bounds__(256)
void final_out_kernel(const float* __restrict__ pooled, const float* __restrict__ a,
                      const float* __restrict__ bb, float* __restrict__ out) {
  const size_t i = (size_t)blockIdx.x * 256 + threadIdx.x;   // < NGROUP*256
  const int c = (int)(i & 255);
  const size_t g = i >> 8;
  const int b = (int)(g >> 10), s = (int)(g & 1023);
  float y = fmaxf(a[c] * pooled[i] + bb[c], 0.0f);
  out[(size_t)NB * 3 * NPOINT + ((size_t)b * 256 + c) * NPOINT + s] = y;
}

// ---------------------------------------------------------------------------
extern "C" void kernel_launch(void* const* d_in, const int* in_sizes, int n_in,
                              void* d_out, int out_size, void* d_ws, size_t ws_size,
                              hipStream_t stream) {
  const float* xyz   = (const float*)d_in[0];
  const float* feats = (const float*)d_in[1];
  const float* W1 = (const float*)d_in[2];
  const float* W2 = (const float*)d_in[3];
  const float* W3 = (const float*)d_in[4];
  // d_in[5..7] = conv biases (cancel under batch-norm; zeros in setup)
  const float* g1 = (const float*)d_in[8];
  const float* g2 = (const float*)d_in[9];
  const float* g3 = (const float*)d_in[10];
  const float* b1 = (const float*)d_in[11];
  const float* b2 = (const float*)d_in[12];
  const float* b3 = (const float*)d_in[13];
  (void)in_sizes; (void)n_in; (void)out_size; (void)ws_size;

  char* ws = (char*)d_ws;
  float*     newxyz  = (float*)(ws + OFF_NEWXYZ);
  int*       ballidx = (int*)(ws + OFF_BALL);
  float*     stats   = (float*)(ws + OFF_STATS);   // l*512 : sum[256], sq[256]
  float*     abuf    = (float*)(ws + OFF_AB);      // l*512 : a[256], b[256]
  _Float16*  w1f     = (_Float16*)(ws + OFF_W1F);
  _Float16*  w2f     = (_Float16*)(ws + OFF_W2F);
  _Float16*  w3f     = (_Float16*)(ws + OFF_W3F);
  float*     pooled  = (float*)(ws + OFF_POOL);
  _Float16*  bufA    = (_Float16*)(ws + OFF_BUFA);
  _Float16*  bufB    = (_Float16*)(ws + OFF_BUFB);
  float*     out     = (float*)d_out;

  const float invM = 1.0f / (float)MROWS;

  zero_stats_kernel<<<6, 256, 0, stream>>>(stats);
  fps_kernel<<<NB, 1024, 0, stream>>>(xyz, newxyz, out);
  ballquery_kernel<<<64, 256, 0, stream>>>(xyz, newxyz, ballidx);
  build_x0_kernel<<<MROWS / 256, 256, 0, stream>>>(xyz, feats, newxyz, ballidx, bufA);

  prep_wfrag_kernel<<<8 * 3, 32, 0, stream>>>(W1, 67, 3, w1f);    // 128 out, K 67->96
  prep_wfrag_kernel<<<8 * 4, 32, 0, stream>>>(W2, 128, 4, w2f);   // 128 out, K 128
  prep_wfrag_kernel<<<16 * 4, 32, 0, stream>>>(W3, 128, 4, w3f);  // 256 out, K 128

  const int gemm_blocks = (MROWS / 16) / 8;       // 4096 (8 waves/block)

  // Layer 1: X0(bufA,[Mx96]) -> bufB [Mx128]
  gemm_bn_kernel<<<gemm_blocks, 256, 0, stream>>>(bufA, 96, 3, w1f, bufB,
                                                  stats + 0, stats + 256);
  bn_final_kernel<<<1, 256, 0, stream>>>(stats + 0, stats + 256, g1, b1,
                                         abuf + 0, abuf + 256, 128, invM);
  bn_apply_kernel<<<(MROWS * 128) / (256 * 8), 256, 0, stream>>>(bufB, abuf + 0, abuf + 256);

  // Layer 2: bufB -> bufA [Mx128]
  gemm_bn_kernel<<<gemm_blocks, 256, 0, stream>>>(bufB, 128, 4, w2f, bufA,
                                                  stats + 512, stats + 768);
  bn_final_kernel<<<1, 256, 0, stream>>>(stats + 512, stats + 768, g2, b2,
                                         abuf + 512, abuf + 768, 128, invM);
  bn_apply_kernel<<<(MROWS * 128) / (256 * 8), 256, 0, stream>>>(bufA, abuf + 512, abuf + 768);

  // Layer 3: bufA -> pooled [Gx256] (fused max-pool), then BN+ReLU+transpose
  gemm3_pool_kernel<<<(NGROUP * 4) / 8, 256, 0, stream>>>(bufA, w3f, pooled,
                                                          stats + 1024, stats + 1280);
  bn_final_kernel<<<1, 256, 0, stream>>>(stats + 1024, stats + 1280, g3, b3,
                                         abuf + 1024, abuf + 1280, 256, invM);
  final_out_kernel<<<(NGROUP * 256) / 256, 256, 0, stream>>>(pooled, abuf + 1024,
                                                             abuf + 1280, out);
}
